// RWKV7Attention_78391743086644
// MI455X (gfx1250) — compile-verified, matrix operand order checked by reading, not run on hardware
//
#include <hip/hip_runtime.h>
#include <hip/hip_bf16.h>
#include <math.h>

// ---------------------------------------------------------------------------
// Problem constants (match reference)
// ---------------------------------------------------------------------------
constexpr int BB = 4;
constexpr int TT = 1024;
constexpr int CC = 1024;
constexpr int HH = 16;
constexpr int NN = 64;
constexpr int MT = BB * TT;               // 4096 rows
constexpr float EPS_GN = 1e-5f * 64.0f;   // 1e-5 * 8^2

typedef __attribute__((ext_vector_type(16))) __bf16 v16bf;
typedef __attribute__((ext_vector_type(8)))  __bf16 v8bf;
typedef __attribute__((ext_vector_type(8)))  float  v8f;

static __device__ __forceinline__ __bf16 f2bf(float f) { return (__bf16)f; }

// ---------------------------------------------------------------------------
// 1) Token-shift + six mixes -> bf16 activations
// ---------------------------------------------------------------------------
__global__ void __launch_bounds__(256)
k_prep(const float* __restrict__ x,
       const float* __restrict__ mr, const float* __restrict__ mw,
       const float* __restrict__ mk, const float* __restrict__ mv,
       const float* __restrict__ ma, const float* __restrict__ mg,
       __bf16* __restrict__ Xr, __bf16* __restrict__ Xw, __bf16* __restrict__ Xk,
       __bf16* __restrict__ Xv, __bf16* __restrict__ Xa, __bf16* __restrict__ Xg)
{
    int idx = blockIdx.x * blockDim.x + threadIdx.x;
    if (idx >= MT * CC) return;
    int c = idx % CC;
    int t = (idx / CC) % TT;
    float xc   = x[idx];
    float prev = (t > 0) ? x[idx - CC] : 0.0f;
    float xx   = prev - xc;
    Xr[idx] = f2bf(xc + xx * mr[c]);
    Xw[idx] = f2bf(xc + xx * mw[c]);
    Xk[idx] = f2bf(xc + xx * mk[c]);
    Xv[idx] = f2bf(xc + xx * mv[c]);
    Xa[idx] = f2bf(xc + xx * ma[c]);
    Xg[idx] = f2bf(xc + xx * mg[c]);
}

// ---------------------------------------------------------------------------
// 2) fp32 -> bf16 weight conversion, optional transpose.
// ---------------------------------------------------------------------------
__global__ void __launch_bounds__(256)
k_cvt(const float* __restrict__ src, __bf16* __restrict__ dst,
      int rows, int cols, int transpose)
{
    int idx = blockIdx.x * blockDim.x + threadIdx.x;
    if (idx >= rows * cols) return;
    int r = idx / cols, c = idx % cols;
    float v = src[idx];
    if (transpose) dst[(size_t)c * rows + r] = f2bf(v);
    else           dst[idx] = f2bf(v);
}

// ---------------------------------------------------------------------------
// 3) bf16 WMMA GEMM: Y[M,N] = act( X[M,K] @ Wt[N,K]^T ), fp32 out (+opt bf16)
//    Block = 128 threads (4 waves). Each wave: 32 rows x (NSUB*16) cols,
//    i.e. 2x NSUB accumulators, 8 WMMAs per 32-K step at NSUB=4.
//    NSUB is a compile-time constant so the K loop has NO scalar branches.
//    K multiple of 32; M multiple of 128.
// ---------------------------------------------------------------------------
union bfrag { v16bf v; v8bf h[2]; };

template<int NSUB>
__global__ void __launch_bounds__(128)
k_gemm(const __bf16* __restrict__ X, const __bf16* __restrict__ Wt,
       float* __restrict__ Y, __bf16* __restrict__ Yb,
       int N, int K, int act, int n0base)
{
    const int lane = threadIdx.x & 31;
    const int wave = threadIdx.x >> 5;
    const int m0   = blockIdx.y * 128 + wave * 32;
    const int n0   = n0base + blockIdx.x * 64;
    const int half = lane >> 4;          // 0: K {0..7,16..23}, 1: K {8..15,24..31}
    const int l15  = lane & 15;
    const int koff = half * 8;

    v8f acc[2][NSUB];
    #pragma unroll
    for (int mi = 0; mi < 2; ++mi)
        #pragma unroll
        for (int ns = 0; ns < NSUB; ++ns)
            acc[mi][ns] = (v8f){};

    const __bf16* aP0 = X + (size_t)(m0 + l15) * K + koff;
    const __bf16* aP1 = X + (size_t)(m0 + 16 + l15) * K + koff;
    const __bf16* bP[NSUB];
    #pragma unroll
    for (int ns = 0; ns < NSUB; ++ns)
        bP[ns] = Wt + (size_t)(n0 + ns * 16 + l15) * K + koff;

    for (int k = 0; k < K; k += 32) {
        bfrag a0, a1, bf_[NSUB];
        a0.h[0] = *(const v8bf*)(aP0 + k);
        a0.h[1] = *(const v8bf*)(aP0 + k + 16);
        a1.h[0] = *(const v8bf*)(aP1 + k);
        a1.h[1] = *(const v8bf*)(aP1 + k + 16);
        #pragma unroll
        for (int ns = 0; ns < NSUB; ++ns) {
            bf_[ns].h[0] = *(const v8bf*)(bP[ns] + k);
            bf_[ns].h[1] = *(const v8bf*)(bP[ns] + k + 16);
        }
        __builtin_prefetch(aP0 + k + 256, 0, 3);   // global_prefetch_b8
        #pragma unroll
        for (int ns = 0; ns < NSUB; ++ns) {
            acc[0][ns] = __builtin_amdgcn_wmma_f32_16x16x32_bf16(
                false, a0.v, false, bf_[ns].v, (short)0, acc[0][ns], false, false);
            acc[1][ns] = __builtin_amdgcn_wmma_f32_16x16x32_bf16(
                false, a1.v, false, bf_[ns].v, (short)0, acc[1][ns], false, false);
        }
    }

    #pragma unroll
    for (int mi = 0; mi < 2; ++mi) {
        #pragma unroll
        for (int ns = 0; ns < NSUB; ++ns) {
            int col = n0 + ns * 16 + l15;
            #pragma unroll
            for (int e = 0; e < 8; ++e) {
                int row = m0 + mi * 16 + e + half * 8;
                float v = acc[mi][ns][e];
                if (act == 1)      v = tanhf(v);
                else if (act == 2) v = 1.0f / (1.0f + __expf(-v));
                Y[(size_t)row * N + col] = v;
                if (Yb) Yb[(size_t)row * N + col] = f2bf(v);
            }
        }
    }
}

// ---------------------------------------------------------------------------
// 4) Elementwise epilogues for LoRA paths
// ---------------------------------------------------------------------------
__global__ void __launch_bounds__(256)
k_w_decay(const float* __restrict__ L, const float* __restrict__ w0,
          float* __restrict__ decay)
{
    int idx = blockIdx.x * blockDim.x + threadIdx.x;
    if (idx >= MT * CC) return;
    int c = idx % CC;
    float y = w0[c] + L[idx];
    float w = -log1pf(__expf(-y)) - 0.5f;   // -softplus(-y) - 0.5
    decay[idx] = __expf(-__expf(w));        // exp(-exp(w)) in (0,1)
}

__global__ void __launch_bounds__(256)
k_a_sig(const float* __restrict__ L, const float* __restrict__ a0,
        float* __restrict__ a)
{
    int idx = blockIdx.x * blockDim.x + threadIdx.x;
    if (idx >= MT * CC) return;
    int c = idx % CC;
    a[idx] = 1.0f / (1.0f + __expf(-(a0[c] + L[idx])));
}

__global__ void __launch_bounds__(256)
k_vmix(float* __restrict__ v, const float* __restrict__ v_first,
       const float* __restrict__ v0, const float* __restrict__ L)
{
    int idx = blockIdx.x * blockDim.x + threadIdx.x;
    if (idx >= MT * CC) return;
    int c = idx % CC;
    float s = 1.0f / (1.0f + __expf(-(v0[c] + L[idx])));
    float vv = v[idx];
    v[idx] = vv + (v_first[idx] - vv) * s;
}

// ---------------------------------------------------------------------------
// 5) kk normalization + k update.  One 64-thread block per (b,t,h).
// ---------------------------------------------------------------------------
__device__ __forceinline__ float block_reduce64(float val, float* red, int n)
{
    red[n] = val;
    __syncthreads();
    #pragma unroll
    for (int s = 32; s > 0; s >>= 1) {
        if (n < s) red[n] += red[n + s];
        __syncthreads();
    }
    float r = red[0];
    __syncthreads();
    return r;
}

__global__ void __launch_bounds__(64)
k_kk(const float* __restrict__ k_in, const float* __restrict__ k_k,
     const float* __restrict__ k_a, const float* __restrict__ a,
     float* __restrict__ kk_out, float* __restrict__ k_out)
{
    __shared__ float red[64];
    int n   = threadIdx.x;
    int h   = blockIdx.x % HH;
    int row = blockIdx.x / HH;          // b*T + t
    int c   = h * NN + n;
    size_t idx = (size_t)row * CC + c;

    float kv = k_in[idx];
    float x  = kv * k_k[c];
    float ss = block_reduce64(x * x, red, n);
    float inv = 1.0f / fmaxf(sqrtf(ss), 1e-12f);
    kk_out[idx] = x * inv;
    float av = a[idx];
    k_out[idx] = kv * (1.0f + (av - 1.0f) * k_a[c]);
}

// ---------------------------------------------------------------------------
// 6) wkv7 sequential scan.  One block (64 threads = 2 waves) per (b,h).
//    Thread i owns row i of the 64x64 state in registers.
// ---------------------------------------------------------------------------
__global__ void __launch_bounds__(64)
k_scan(const float* __restrict__ r,   const float* __restrict__ decay,
       const float* __restrict__ k,   const float* __restrict__ v,
       const float* __restrict__ kk,  const float* __restrict__ a,
       float* __restrict__ o)
{
    __shared__ float sr[64], sw[64], sk[64], sv[64], sb[64], sma[64];
    const int i = threadIdx.x;
    const int b = blockIdx.x / HH;
    const int h = blockIdx.x % HH;
    const size_t base = (size_t)b * TT * CC + h * NN + i;

    float S[64];
    #pragma unroll
    for (int j = 0; j < 64; ++j) S[j] = 0.0f;

    for (int t = 0; t < TT; ++t) {
        size_t idx = base + (size_t)t * CC;
        float kkv = kk[idx];
        sr[i]  = r[idx];
        sw[i]  = decay[idx];
        sk[i]  = k[idx];
        sv[i]  = v[idx];
        sb[i]  = kkv * a[idx];   // b vector = kk * a
        sma[i] = -kkv;           // a vector = -kk
        __syncthreads();

        float sa = 0.0f;
        #pragma unroll
        for (int j = 0; j < 64; ++j) sa += S[j] * sma[j];

        float vi = sv[i];
        float y  = 0.0f;
        #pragma unroll
        for (int j = 0; j < 64; ++j) {
            S[j] = S[j] * sw[j] + sa * sb[j] + vi * sk[j];
            y += S[j] * sr[j];
        }
        o[idx] = y;
        __syncthreads();
    }
}

// ---------------------------------------------------------------------------
// 7) GroupNorm + bonus + gate -> bf16 for final projection.
// ---------------------------------------------------------------------------
__global__ void __launch_bounds__(64)
k_post(const float* __restrict__ o, const float* __restrict__ r,
       const float* __restrict__ k, const float* __restrict__ v,
       const float* __restrict__ g, const float* __restrict__ r_k,
       const float* __restrict__ ln_g, const float* __restrict__ ln_b,
       __bf16* __restrict__ gated)
{
    __shared__ float red[64];
    int n   = threadIdx.x;
    int h   = blockIdx.x % HH;
    int row = blockIdx.x / HH;
    int c   = h * NN + n;
    size_t idx = (size_t)row * CC + c;

    float ov   = o[idx];
    float mean = block_reduce64(ov, red, n) * (1.0f / 64.0f);
    float d    = ov - mean;
    float var  = block_reduce64(d * d, red, n) * (1.0f / 64.0f);
    float y    = d * rsqrtf(var + EPS_GN) * ln_g[c] + ln_b[c];

    float s = block_reduce64(r[idx] * k[idx] * r_k[c], red, n);
    float outv = (y + s * v[idx]) * g[idx];
    gated[idx] = f2bf(outv);
}

// ---------------------------------------------------------------------------
// Host-side orchestration
// ---------------------------------------------------------------------------
extern "C" void kernel_launch(void* const* d_in, const int* in_sizes, int n_in,
                              void* d_out, int out_size, void* d_ws, size_t ws_size,
                              hipStream_t stream)
{
    (void)in_sizes; (void)n_in; (void)out_size; (void)ws_size;

    const float* x       = (const float*)d_in[0];
    const float* v_first = (const float*)d_in[1];
    const float* x_r = (const float*)d_in[2];
    const float* x_w = (const float*)d_in[3];
    const float* x_k = (const float*)d_in[4];
    const float* x_v = (const float*)d_in[5];
    const float* x_a = (const float*)d_in[6];
    const float* x_g = (const float*)d_in[7];
    const float* w0  = (const float*)d_in[8];
    const float* w1  = (const float*)d_in[9];
    const float* w2  = (const float*)d_in[10];
    const float* a0  = (const float*)d_in[11];
    const float* a1  = (const float*)d_in[12];
    const float* a2  = (const float*)d_in[13];
    const float* v0  = (const float*)d_in[14];
    const float* v1  = (const float*)d_in[15];
    const float* v2  = (const float*)d_in[16];
    const float* g1  = (const float*)d_in[17];
    const float* g2  = (const float*)d_in[18];
    const float* k_k = (const float*)d_in[19];
    const float* k_a = (const float*)d_in[20];
    const float* r_k = (const float*)d_in[21];
    const float* W_r = (const float*)d_in[22];
    const float* W_k = (const float*)d_in[23];
    const float* W_v = (const float*)d_in[24];
    const float* W_o = (const float*)d_in[25];
    const float* ln_g = (const float*)d_in[26];
    const float* ln_b = (const float*)d_in[27];

    // ---- workspace bump allocator ----
    char* ws = (char*)d_ws;
    size_t off = 0;
    auto alloc = [&](size_t bytes) -> void* {
        void* p = ws + off;
        off = (off + bytes + 255) & ~(size_t)255;
        return p;
    };
    const size_t AB = (size_t)MT * CC * sizeof(__bf16);   // bf16 activation
    const size_t AF = (size_t)MT * CC * sizeof(float);    // fp32 activation
    __bf16* Xr = (__bf16*)alloc(AB);
    __bf16* Xw = (__bf16*)alloc(AB);
    __bf16* Xk = (__bf16*)alloc(AB);
    __bf16* Xv = (__bf16*)alloc(AB);
    __bf16* Xa = (__bf16*)alloc(AB);
    __bf16* Xg = (__bf16*)alloc(AB);
    __bf16* Wr_b = (__bf16*)alloc((size_t)CC * CC * 2);
    __bf16* Wk_b = (__bf16*)alloc((size_t)CC * CC * 2);
    __bf16* Wv_b = (__bf16*)alloc((size_t)CC * CC * 2);
    __bf16* Wo_b = (__bf16*)alloc((size_t)CC * CC * 2);
    __bf16* w1t = (__bf16*)alloc((size_t)64  * CC * 2);
    __bf16* w2t = (__bf16*)alloc((size_t)CC  * 64 * 2);
    __bf16* a1t = (__bf16*)alloc((size_t)64  * CC * 2);
    __bf16* a2t = (__bf16*)alloc((size_t)CC  * 64 * 2);
    __bf16* v1t = (__bf16*)alloc((size_t)32  * CC * 2);
    __bf16* v2t = (__bf16*)alloc((size_t)CC  * 32 * 2);
    __bf16* g1t = (__bf16*)alloc((size_t)160 * CC * 2);
    __bf16* g2t = (__bf16*)alloc((size_t)CC  * 160 * 2);
    float* rbuf   = (float*)alloc(AF);
    float* kbuf   = (float*)alloc(AF);
    float* vbuf   = (float*)alloc(AF);
    float* decayb = (float*)alloc(AF);
    float* abuf   = (float*)alloc(AF);
    float* gbuf   = (float*)alloc(AF);
    float* kkbuf  = (float*)alloc(AF);
    float* Lbuf   = (float*)alloc(AF);
    float* obuf   = (float*)alloc(AF);
    float*  tmpf = (float*) alloc((size_t)MT * 160 * sizeof(float));
    __bf16* tmpb = (__bf16*)alloc((size_t)MT * 160 * sizeof(__bf16));
    __bf16* gated = (__bf16*)alloc(AB);

    const int EW = (MT * CC + 255) / 256;   // elementwise grid

    // 1) token shift + mixes
    k_prep<<<EW, 256, 0, stream>>>(x, x_r, x_w, x_k, x_v, x_a, x_g,
                                   Xr, Xw, Xk, Xv, Xa, Xg);

    // 2) weight conversions
    auto cvt = [&](const float* s, __bf16* d, int r_, int c_, int tr) {
        k_cvt<<<(r_ * c_ + 255) / 256, 256, 0, stream>>>(s, d, r_, c_, tr);
    };
    cvt(W_r, Wr_b, CC, CC, 0);
    cvt(W_k, Wk_b, CC, CC, 0);
    cvt(W_v, Wv_b, CC, CC, 0);
    cvt(W_o, Wo_b, CC, CC, 0);
    cvt(w1, w1t, CC, 64, 1);   cvt(w2, w2t, 64, CC, 1);
    cvt(a1, a1t, CC, 64, 1);   cvt(a2, a2t, 64, CC, 1);
    cvt(v1, v1t, CC, 32, 1);   cvt(v2, v2t, 32, CC, 1);
    cvt(g1, g1t, CC, 160, 1);  cvt(g2, g2t, 160, CC, 1);

    // GEMM dispatch: full 64-wide tiles with NSUB=4, then 16/32/48 tail.
    auto gemm = [&](const __bf16* X_, const __bf16* Wt_, float* Y_, __bf16* Yb_,
                    int M_, int N_, int K_, int act_) {
        int nfull = N_ / 64;
        int rem   = N_ - nfull * 64;
        dim3 blk(128);
        if (nfull > 0) {
            dim3 grid(nfull, M_ / 128);
            k_gemm<4><<<grid, blk, 0, stream>>>(X_, Wt_, Y_, Yb_, N_, K_, act_, 0);
        }
        if (rem > 0) {
            dim3 grid(1, M_ / 128);
            int n0base = nfull * 64;
            if (rem == 16)
                k_gemm<1><<<grid, blk, 0, stream>>>(X_, Wt_, Y_, Yb_, N_, K_, act_, n0base);
            else if (rem == 32)
                k_gemm<2><<<grid, blk, 0, stream>>>(X_, Wt_, Y_, Yb_, N_, K_, act_, n0base);
            else
                k_gemm<3><<<grid, blk, 0, stream>>>(X_, Wt_, Y_, Yb_, N_, K_, act_, n0base);
        }
    };

    // 3) big projections
    gemm(Xr, Wr_b, rbuf, nullptr, MT, CC, CC, 0);
    gemm(Xk, Wk_b, kbuf, nullptr, MT, CC, CC, 0);
    gemm(Xv, Wv_b, vbuf, nullptr, MT, CC, CC, 0);

    // 4) decay LoRA: w = -softplus(-(w0 + tanh(xw@w1)@w2)) - 0.5; decay=exp(-exp(w))
    gemm(Xw, w1t, tmpf, tmpb, MT, 64, CC, 1);          // tanh epilogue
    gemm(tmpb, w2t, Lbuf, nullptr, MT, CC, 64, 0);
    k_w_decay<<<EW, 256, 0, stream>>>(Lbuf, w0, decayb);

    // 5) a LoRA: a = sigmoid(a0 + (xa@a1)@a2)
    gemm(Xa, a1t, tmpf, tmpb, MT, 64, CC, 0);
    gemm(tmpb, a2t, Lbuf, nullptr, MT, CC, 64, 0);
    k_a_sig<<<EW, 256, 0, stream>>>(Lbuf, a0, abuf);

    // 6) v mix toward v_first
    gemm(Xv, v1t, tmpf, tmpb, MT, 32, CC, 0);
    gemm(tmpb, v2t, Lbuf, nullptr, MT, CC, 32, 0);
    k_vmix<<<EW, 256, 0, stream>>>(vbuf, v_first, v0, Lbuf);

    // 7) gate: g = sigmoid(xg@g1) @ g2
    gemm(Xg, g1t, tmpf, tmpb, MT, 160, CC, 2);         // sigmoid epilogue
    gemm(tmpb, g2t, gbuf, nullptr, MT, CC, 160, 0);

    // 8) kk normalization + k update (in place on kbuf)
    k_kk<<<MT * HH, 64, 0, stream>>>(kbuf, k_k, k_a, abuf, kkbuf, kbuf);

    // 9) wkv7 scan: 64 independent (b,h) state recurrences
    k_scan<<<BB * HH, 64, 0, stream>>>(rbuf, decayb, kbuf, vbuf, kkbuf, abuf, obuf);

    // 10) GroupNorm + bonus + gate -> bf16
    k_post<<<MT * HH, 64, 0, stream>>>(obuf, rbuf, kbuf, vbuf, gbuf,
                                       r_k, ln_g, ln_b, gated);

    // 11) final projection straight into d_out (first tuple element, fp32)
    gemm(gated, Wo_b, (float*)d_out, nullptr, MT, CC, CC, 0);

    // 12) second tuple element: v_first passthrough
    hipMemcpyAsync((float*)d_out + (size_t)MT * CC, v_first,
                   (size_t)MT * CC * sizeof(float),
                   hipMemcpyDeviceToDevice, stream);
}
